// PCT_semseg_1984274891061
// MI455X (gfx1250) — matrix operand order repaired and checked by reading, not compiled, not run on hardware
//
#include <hip/hip_runtime.h>
#include <hip/hip_bf16.h>

// ===========================================================================
// PCT semantic-segmentation forward for MI455X (gfx1250, wave32, WMMA).
//   - All 1x1 convs + attention GEMMs run through v_wmma_f32_16x16x32_f16.
//   - GEMM X tiles double-buffered via global_load_async_to_lds_b128
//     (per-wave ASYNCcnt: wave 0 stages, workgroup barrier publishes).
//   - Attention is flash-style: never materializes the 4096x4096 matrix.
// ===========================================================================

typedef __attribute__((ext_vector_type(16))) _Float16 v16h;
typedef __attribute__((ext_vector_type(8)))  _Float16 v8h;
typedef __attribute__((ext_vector_type(8)))  float    v8f;

#define NPTS  4096
#define BATCH 4
#define KNN   20
#define NKI   (NPTS * KNN)   // 81920

// -------------------------- CDNA5 async helpers ----------------------------
__device__ __forceinline__ unsigned lds_off(const void* p) {
  // Flat LDS addresses carry the LDS byte offset in the low 32 bits.
  return (unsigned)(size_t)p;
}
__device__ __forceinline__ void async_b128(unsigned lds, unsigned long long ga) {
  asm volatile("global_load_async_to_lds_b128 %0, %1, off" :: "v"(lds), "v"(ga) : "memory");
}
__device__ __forceinline__ void async_b128_off16(unsigned lds, unsigned long long ga) {
  // INST_OFFSET is added to BOTH the LDS and the global address (ISA 08 §4.4)
  asm volatile("global_load_async_to_lds_b128 %0, %1, off offset:16" :: "v"(lds), "v"(ga) : "memory");
}
__device__ __forceinline__ void async_wait0() {
  asm volatile("s_wait_asynccnt 0" ::: "memory");
}

__device__ __forceinline__ v8f wmma16(v16h a, v16h b, v8f c) {
  return __builtin_amdgcn_wmma_f32_16x16x32_f16(false, a, false, b, (short)0, c, false, false);
}

// ------------------------------ f32 -> f16 ---------------------------------
__global__ void f2h(const float* __restrict__ s, _Float16* __restrict__ d, long n) {
  long i = (long)blockIdx.x * blockDim.x + threadIdx.x;
  if (i < n) d[i] = (_Float16)s[i];
}

// ------------------------------- KNN top-20 --------------------------------
__global__ void knn_topk(const float* __restrict__ x, int* __restrict__ idx) {
  __shared__ float tx[3][256];
  int gid = blockIdx.x * 256 + threadIdx.x;       // blocks never straddle batches
  int b = gid / NPTS, n = gid % NPTS;
  const float* xb = x + (long)b * 3 * NPTS;
  float px = xb[n], py = xb[NPTS + n], pz = xb[2 * NPTS + n];
  float bd[KNN]; int bi[KNN];
  #pragma unroll
  for (int j = 0; j < KNN; j++) { bd[j] = -1e30f; bi[j] = 0; }
  for (int t0 = 0; t0 < NPTS; t0 += 256) {
    __syncthreads();
    tx[0][threadIdx.x] = xb[t0 + threadIdx.x];
    tx[1][threadIdx.x] = xb[NPTS + t0 + threadIdx.x];
    tx[2][threadIdx.x] = xb[2 * NPTS + t0 + threadIdx.x];
    __syncthreads();
    for (int j = 0; j < 256; j++) {
      float dx = px - tx[0][j], dy = py - tx[1][j], dz = pz - tx[2][j];
      float d = -(dx * dx + dy * dy + dz * dz);
      if (d > bd[KNN - 1]) {
        int p = KNN - 1;
        while (p > 0 && bd[p - 1] < d) { bd[p] = bd[p - 1]; bi[p] = bi[p - 1]; p--; }
        bd[p] = d; bi[p] = t0 + j;
      }
    }
  }
  int* op = idx + (long)gid * KNN;
  for (int j = 0; j < KNN; j++) op[j] = bi[j];
}

// ------------------- edge features + conv1 (64x6, scalar) ------------------
__global__ void edge_conv1(const float* __restrict__ x, const int* __restrict__ idx,
                           const float* __restrict__ w, _Float16* __restrict__ e1) {
  __shared__ float feat[6][KNN];
  int bn = blockIdx.x;
  int b = bn / NPTS, n = bn % NPTS;
  int tid = threadIdx.x;                     // 64 threads
  if (tid < KNN) {
    int m = idx[((long)b * NPTS + n) * KNN + tid];
    #pragma unroll
    for (int c = 0; c < 3; c++) {
      float ctr = x[((long)b * 3 + c) * NPTS + n];
      float nb  = x[((long)b * 3 + c) * NPTS + m];
      feat[c][tid]     = nb - ctr;
      feat[c + 3][tid] = ctr;
    }
  }
  __syncthreads();
  float wr[6];
  #pragma unroll
  for (int c = 0; c < 6; c++) wr[c] = w[tid * 6 + c];
  long base = (((long)b * 64 + tid) * NPTS + n) * (long)KNN;
  for (int k = 0; k < KNN; k++) {
    float s = 0.f;
    #pragma unroll
    for (int c = 0; c < 6; c++) s += wr[c] * feat[c][k];
    e1[base + k] = (_Float16)s;
  }
}

// --------------------- per-channel batch statistics ------------------------
__global__ void chan_stats(const _Float16* __restrict__ y, float* __restrict__ sums,
                           int C, long inner) {
  __shared__ float rs[256], rq[256];
  int c = blockIdx.x;
  long per = (long)BATCH * inner;
  float s = 0.f, s2 = 0.f;
  for (long i = threadIdx.x; i < per; i += 256) {
    long b = i / inner, r = i - b * inner;
    float v = (float)y[((long)b * C + c) * inner + r];
    s += v; s2 += v * v;
  }
  rs[threadIdx.x] = s; rq[threadIdx.x] = s2;
  __syncthreads();
  for (int o = 128; o > 0; o >>= 1) {
    if (threadIdx.x < o) { rs[threadIdx.x] += rs[threadIdx.x + o]; rq[threadIdx.x] += rq[threadIdx.x + o]; }
    __syncthreads();
  }
  if (threadIdx.x == 0) { sums[2 * c] = rs[0]; sums[2 * c + 1] = rq[0]; }
}

// --------------------- in-place batchnorm + activation ---------------------
// act: 0=none, 1=leaky(0.2), 2=relu
__global__ void bn_act(_Float16* __restrict__ y, const float* __restrict__ sums,
                       const float* __restrict__ g, const float* __restrict__ bt,
                       int C, long inner, long total, float invcnt, int act) {
  long i = (long)blockIdx.x * blockDim.x + threadIdx.x;
  if (i >= total) return;
  long t = i / inner; int c = (int)(t % C);
  float mean = sums[2 * c] * invcnt;
  float var  = sums[2 * c + 1] * invcnt - mean * mean;
  float v = ((float)y[i] - mean) * rsqrtf(var + 1e-5f) * g[c] + bt[c];
  if (act == 1) v = v > 0.f ? v : 0.2f * v;
  else if (act == 2) v = fmaxf(v, 0.f);
  y[i] = (_Float16)v;
}

// ----------------- bn + leaky + max over k (for edge conv2) ----------------
__global__ void bn_leaky_maxk(const _Float16* __restrict__ y, const float* __restrict__ sums,
                              const float* __restrict__ g, const float* __restrict__ bt,
                              _Float16* __restrict__ out) {
  int i = blockIdx.x * blockDim.x + threadIdx.x;   // over B*64*N
  if (i >= BATCH * 64 * NPTS) return;
  int c = (i / NPTS) % 64;
  float invc = 1.f / (float)((long)BATCH * NKI);
  float mean = sums[2 * c] * invc;
  float var  = sums[2 * c + 1] * invc - mean * mean;
  float sc = rsqrtf(var + 1e-5f) * g[c], sh = bt[c];
  const _Float16* p = y + (long)i * KNN;
  float m = -1e30f;
  for (int k = 0; k < KNN; k++) {
    float v = ((float)p[k] - mean) * sc + sh;
    v = v > 0.f ? v : 0.2f * v;
    m = fmaxf(m, v);
  }
  out[i] = (_Float16)m;
}

// ------------------------- generic WMMA GEMM -------------------------------
// Y(B,M,inner) = W[MxKd] * X(B,Kd,inner) (+bias). fp16 in, f32 acc.
// 4 waves / block: wave w owns M rows [by*64+16w, +16); one 16-col tile.
// X tile (32k x 16col) double-buffered in LDS via async-to-LDS (wave 0 only;
// ASYNCcnt is per-wave, the workgroup barrier publishes the tile).
template <bool HASBIAS, bool F32OUT>
__global__ void wmma_gemm_t(const _Float16* __restrict__ W, const _Float16* __restrict__ X,
                            const float* __restrict__ bias, void* __restrict__ Yv,
                            int M, int Kd, long inner) {
  __shared__ _Float16 lx[2][32][16];               // [buf][k][col]
  int tid = threadIdx.x;                           // 128
  int wave = tid >> 5, lane = tid & 31;
  int hb = lane >> 4, l = lane & 15;
  long col0 = (long)blockIdx.x * 16;
  int  m0   = blockIdx.y * 64 + wave * 16;
  long b  = col0 / inner;                          // inner%16==0 -> tile in one batch
  long n0 = col0 - b * inner;
  const _Float16* Xb = X + (long)b * Kd * inner + n0;
  int mrow = m0 + l; if (mrow >= M) mrow = M - 1;  // clamp; stores are guarded
  const _Float16* Wr = W + (long)mrow * Kd;
  int steps = Kd >> 5;

  if (wave == 0) {                                 // prologue: stage k-tile 0
    unsigned ld = lds_off(&lx[0][lane][0]);
    unsigned long long ga = (unsigned long long)(size_t)(Xb + (long)lane * inner);
    async_b128(ld, ga);
    async_b128_off16(ld, ga);
  }

  v8f acc = {};
  for (int i = 0; i < steps; i++) {
    if (wave == 0) async_wait0();                  // tile i landed in LDS
    __syncthreads();
    int buf = i & 1;
    if (wave == 0 && i + 1 < steps) {              // prefetch tile i+1 (overlaps WMMA)
      unsigned ld = lds_off(&lx[buf ^ 1][lane][0]);
      unsigned long long ga =
          (unsigned long long)(size_t)(Xb + (long)((i + 1) * 32 + lane) * inner);
      async_b128(ld, ga);
      async_b128_off16(ld, ga);
    }
    int k0 = i << 5;
    const _Float16* wp = Wr + k0 + hb * 8;
    __builtin_prefetch((const void*)(wp + 32), 0, 3);   // global_prefetch_b8: next W frag
    v8h lo = *(const v8h*)wp;
    v8h hi = *(const v8h*)(wp + 16);
    v16h a = __builtin_shufflevector(lo, hi, 0,1,2,3,4,5,6,7,8,9,10,11,12,13,14,15);
    v16h bf = {};
    #pragma unroll
    for (int e = 0; e < 16; e++) bf[e] = lx[buf][8 * hb + e + (e >= 8 ? 8 : 0)][l];
    acc = wmma16(a, bf, acc);
    __syncthreads();                               // all reads of buf done before reuse
  }

  #pragma unroll
  for (int r = 0; r < 8; r++) {
    int m = m0 + r + hb * 8;
    if (m < M) {
      float v = acc[r];
      if (HASBIAS) v += bias[m];
      long o = ((long)b * M + m) * inner + n0 + l;
      if (F32OUT) ((float*)Yv)[o] = v;
      else        ((_Float16*)Yv)[o] = (_Float16)v;
    }
  }
}

// ------------------- attention pass 1: row max / 1/rowsum ------------------
// scores s[n,m] = sum_o qk[o,n]*qk[o,m] (Gram matrix, q and k share weights)
__global__ void attn_rowstats(const _Float16* __restrict__ qk,
                              float* __restrict__ rowmax, float* __restrict__ rowrcp) {
  int wid  = blockIdx.x * 8 + (threadIdx.x >> 5);   // one wave per row, B*N rows
  int lane = threadIdx.x & 31;
  int b = wid / NPTS, n = wid % NPTS;
  const _Float16* Q = qk + (long)b * 32 * NPTS;
  float qn[32];
  #pragma unroll
  for (int o = 0; o < 32; o++) qn[o] = (float)Q[(long)o * NPTS + n];
  float mx = -1e30f, sm = 0.f;
  for (int m0 = 0; m0 < NPTS; m0 += 32) {
    int m = m0 + lane;
    float s = 0.f;
    #pragma unroll
    for (int o = 0; o < 32; o++) s += qn[o] * (float)Q[(long)o * NPTS + m];
    float nm = fmaxf(mx, s);
    sm = sm * __expf(mx - nm) + __expf(s - nm);
    mx = nm;
  }
  for (int off = 16; off > 0; off >>= 1) {
    float omx = __shfl_xor(mx, off, 32);
    float osm = __shfl_xor(sm, off, 32);
    float nm = fmaxf(mx, omx);
    sm = sm * __expf(mx - nm) + osm * __expf(omx - nm);
    mx = nm;
  }
  if (lane == 0) { rowmax[wid] = mx; rowrcp[wid] = 1.f / sm; }
}

// ------------- attention pass 2: streaming colsum + x_r (WMMA) -------------
// xr[c,m] = (Σ_n v[c,n]·p[n,m]) / (1e-9 + Σ_n p[n,m]),  p = exp(s-max)/rowsum
__global__ void attn_xr(const _Float16* __restrict__ qk, const _Float16* __restrict__ v,
                        const float* __restrict__ rowmax, const float* __restrict__ rowrcp,
                        _Float16* __restrict__ xr) {
  __shared__ _Float16 Pl[32][17];
  __shared__ float colsumL[16];
  int b  = blockIdx.y;
  int m0 = blockIdx.x * 16;
  int tid = threadIdx.x, wave = tid >> 5, lane = tid & 31;
  int hb = lane >> 4, l = lane & 15;
  if (tid < 16) colsumL[tid] = 0.f;
  __syncthreads();
  const _Float16* Qb = qk + (long)b * 32 * NPTS;
  const _Float16* Vb = v  + (long)b * 128 * NPTS;
  // loop-invariant B-fragment of K matrix (K=o, cols m0..m0+15)
  v16h bK = {};
  #pragma unroll
  for (int e = 0; e < 16; e++) bK[e] = Qb[(long)(8 * hb + e + (e >= 8 ? 8 : 0)) * NPTS + m0 + l];
  v8f acc0 = {}, acc1 = {};
  int c0 = wave * 32;
  for (int n0 = 0; n0 < NPTS; n0 += 32) {
    if (wave < 2) {                               // compute P tile (exp scores)
      int nr0 = n0 + wave * 16;
      v16h aQ = {};
      #pragma unroll
      for (int e = 0; e < 8; e++) {               // A: M=n, K=o
        aQ[e]     = Qb[(long)(hb * 8 + e) * NPTS + nr0 + l];
        aQ[e + 8] = Qb[(long)(hb * 8 + 16 + e) * NPTS + nr0 + l];
      }
      v8f sf = {};
      sf = wmma16(aQ, bK, sf);
      float csum = 0.f;
      #pragma unroll
      for (int r = 0; r < 8; r++) {
        int nn = nr0 + r + hb * 8;
        float p = __expf(sf[r] - rowmax[(long)b * NPTS + nn]) * rowrcp[(long)b * NPTS + nn];
        csum += p;
        Pl[wave * 16 + r + hb * 8][l] = (_Float16)p;
      }
      atomicAdd(&colsumL[l], csum);               // ds_add_f32
    }
    __syncthreads();
    // B-fragment of P (K = n-local)
    v16h bP = {};
    #pragma unroll
    for (int e = 0; e < 16; e++) bP[e] = Pl[8 * hb + e + (e >= 8 ? 8 : 0)][l];
    // A-fragments of V (rows c, K = n)
    const _Float16* vp0 = Vb + (long)(c0 + l) * NPTS + n0 + hb * 8;
    v8h l0 = *(const v8h*)vp0;
    v8h h0 = *(const v8h*)(vp0 + 16);
    v16h a0 = __builtin_shufflevector(l0, h0, 0,1,2,3,4,5,6,7,8,9,10,11,12,13,14,15);
    acc0 = wmma16(a0, bP, acc0);
    const _Float16* vp1 = vp0 + (long)16 * NPTS;
    v8h l1 = *(const v8h*)vp1;
    v8h h1 = *(const v8h*)(vp1 + 16);
    v16h a1 = __builtin_shufflevector(l1, h1, 0,1,2,3,4,5,6,7,8,9,10,11,12,13,14,15);
    acc1 = wmma16(a1, bP, acc1);
    __syncthreads();
  }
  float cs = 1e-9f + colsumL[l];
  #pragma unroll
  for (int r = 0; r < 8; r++) {
    int c  = c0 + r + hb * 8;
    int c2 = c0 + 16 + r + hb * 8;
    long m = m0 + l;
    xr[((long)b * 128 + c)  * NPTS + m] = (_Float16)(acc0[r] / cs);
    xr[((long)b * 128 + c2) * NPTS + m] = (_Float16)(acc1[r] / cs);
  }
}

// ------------------------------ elementwise --------------------------------
__global__ void sub_k(const _Float16* __restrict__ a, const _Float16* __restrict__ b,
                      _Float16* __restrict__ o, long n) {
  long i = (long)blockIdx.x * blockDim.x + threadIdx.x;
  if (i < n) o[i] = (_Float16)((float)a[i] - (float)b[i]);
}

__global__ void sa_update(_Float16* __restrict__ h, const _Float16* __restrict__ y,
                          _Float16* __restrict__ cat, int layer) {
  long i = (long)blockIdx.x * blockDim.x + threadIdx.x;
  if (i >= (long)BATCH * 128 * NPTS) return;
  int n = (int)(i % NPTS);
  long t = i / NPTS;
  int c = (int)(t % 128), b = (int)(t / 128);
  float nv = (float)h[i] + (float)y[i];
  h[i] = (_Float16)nv;
  cat[((long)b * 512 + layer * 128 + c) * NPTS + n] = (_Float16)nv;
}

__global__ void pool_maxmean(const _Float16* __restrict__ f, float* __restrict__ pooled) {
  __shared__ float smax[256], ssum[256];
  int bc = blockIdx.x;                              // B*1024
  const _Float16* p = f + (long)bc * NPTS;
  float mx = -1e30f, s = 0.f;
  for (int i = threadIdx.x; i < NPTS; i += 256) { float v = (float)p[i]; mx = fmaxf(mx, v); s += v; }
  smax[threadIdx.x] = mx; ssum[threadIdx.x] = s;
  __syncthreads();
  for (int o = 128; o > 0; o >>= 1) {
    if (threadIdx.x < o) {
      smax[threadIdx.x] = fmaxf(smax[threadIdx.x], smax[threadIdx.x + o]);
      ssum[threadIdx.x] += ssum[threadIdx.x + o];
    }
    __syncthreads();
  }
  if (threadIdx.x == 0) { pooled[bc * 2] = smax[0]; pooled[bc * 2 + 1] = ssum[0] / (float)NPTS; }
}

__global__ void build_cat2(const _Float16* __restrict__ f4, const float* __restrict__ pooled,
                           _Float16* __restrict__ cat2) {
  long i = (long)blockIdx.x * blockDim.x + threadIdx.x;   // B*3072*N
  if (i >= (long)BATCH * 3072 * NPTS) return;
  int n = (int)(i % NPTS);
  long t = i / NPTS;
  int c = (int)(t % 3072), b = (int)(t / 3072);
  _Float16 v;
  if (c < 1024)      v = f4[((long)b * 1024 + c) * NPTS + n];
  else if (c < 2048) v = (_Float16)pooled[(b * 1024 + (c - 1024)) * 2 + 0];
  else               v = (_Float16)pooled[(b * 1024 + (c - 2048)) * 2 + 1];
  cat2[i] = v;
}

// ===========================================================================
extern "C" void kernel_launch(void* const* d_in, const int* in_sizes, int n_in,
                              void* d_out, int out_size, void* d_ws, size_t ws_size,
                              hipStream_t stream) {
  const float* x      = (const float*)d_in[0];
  const float* c1_w   = (const float*)d_in[2];
  const float* bn1_g  = (const float*)d_in[3];
  const float* bn1_b  = (const float*)d_in[4];
  const float* bn2_g  = (const float*)d_in[6];
  const float* bn2_b  = (const float*)d_in[7];
  const float* bn3_g  = (const float*)d_in[9];
  const float* bn3_b  = (const float*)d_in[10];
  const float* sa_vb  = (const float*)d_in[13];
  const float* sa_tb  = (const float*)d_in[15];
  const float* sa_g   = (const float*)d_in[16];
  const float* sa_b   = (const float*)d_in[17];
  const float* bn4_g  = (const float*)d_in[19];
  const float* bn4_b  = (const float*)d_in[20];
  const float* c5_b   = (const float*)d_in[22];
  const float* bn5_g  = (const float*)d_in[23];
  const float* bn5_b  = (const float*)d_in[24];
  const float* c6_b   = (const float*)d_in[26];
  const float* bn6_g  = (const float*)d_in[27];
  const float* bn6_b  = (const float*)d_in[28];
  const float* c7_b   = (const float*)d_in[30];

  char* ws = (char*)d_ws;
  size_t off = 0;
  auto alloc = [&](size_t bytes) { size_t o = off; off += (bytes + 255) & ~(size_t)255; return o; };

  int*      idx   = (int*)     (ws + alloc((size_t)BATCH * NPTS * KNN * 4));
  _Float16* w16   = (_Float16*)(ws + alloc(2389504ull * 2));
  float*    stats = (float*)   (ws + alloc(2 * 1024 * 4));
  size_t bigOff = off;                         // arena reused by cat2 later
  _Float16* e1    = (_Float16*)(ws + alloc(41943040ull));
  _Float16* e2    = (_Float16*)(ws + alloc(41943040ull));
  _Float16* h64   = (_Float16*)(ws + alloc(2097152ull));
  _Float16* h128  = (_Float16*)(ws + alloc(4194304ull));
  _Float16* qk16  = (_Float16*)(ws + alloc(1048576ull));
  _Float16* v16b  = (_Float16*)(ws + alloc(4194304ull));
  float*    rmax  = (float*)   (ws + alloc(65536ull));
  float*    rrcp  = (float*)   (ws + alloc(65536ull));
  _Float16* xr16  = (_Float16*)(ws + alloc(4194304ull));
  _Float16* tmp16 = (_Float16*)(ws + alloc(4194304ull));
  _Float16* y16   = (_Float16*)(ws + alloc(4194304ull));
  _Float16* cat   = (_Float16*)(ws + alloc(16777216ull));
  _Float16* f4    = (_Float16*)(ws + alloc(33554432ull));
  float*    pooled= (float*)   (ws + alloc(32768ull));
  _Float16* f5    = (_Float16*)(ws + alloc(16777216ull));
  _Float16* f6    = (_Float16*)(ws + alloc(8388608ull));
  _Float16* cat2  = (_Float16*)(ws + bigOff);  // 100.7MB <= 108.1MB dead arena

  _Float16* w_c2 = w16 + 0;
  _Float16* w_c3 = w16 + 4096;
  _Float16* w_qk = w16 + 12288;
  _Float16* w_vw = w16 + 28672;
  _Float16* w_tw = w16 + 94208;
  _Float16* w_c4 = w16 + 159744;
  _Float16* w_c5 = w16 + 684032;
  _Float16* w_c6 = w16 + 2256896;
  _Float16* w_c7 = w16 + 2387968;

  auto cvt = [&](const float* s, _Float16* d, long n) {
    f2h<<<(unsigned)((n + 255) / 256), 256, 0, stream>>>(s, d, n);
  };
  cvt((const float*)d_in[5],  w_c2, 4096);
  cvt((const float*)d_in[8],  w_c3, 8192);
  cvt((const float*)d_in[11], w_qk, 16384);
  cvt((const float*)d_in[12], w_vw, 65536);
  cvt((const float*)d_in[14], w_tw, 65536);
  cvt((const float*)d_in[18], w_c4, 524288);
  cvt((const float*)d_in[21], w_c5, 1572864);
  cvt((const float*)d_in[25], w_c6, 131072);
  cvt((const float*)d_in[29], w_c7, 1536);

  auto gemm = [&](const _Float16* W, const _Float16* X, const float* bias,
                  void* Y, bool f32out, int M, int Kd, long inner) {
    dim3 g((unsigned)(((long)BATCH * inner) / 16), (unsigned)((M + 63) / 64));
    if (bias) {
      if (f32out) wmma_gemm_t<true,  true ><<<g, 128, 0, stream>>>(W, X, bias, Y, M, Kd, inner);
      else        wmma_gemm_t<true,  false><<<g, 128, 0, stream>>>(W, X, bias, Y, M, Kd, inner);
    } else {
      if (f32out) wmma_gemm_t<false, true ><<<g, 128, 0, stream>>>(W, X, bias, Y, M, Kd, inner);
      else        wmma_gemm_t<false, false><<<g, 128, 0, stream>>>(W, X, bias, Y, M, Kd, inner);
    }
  };
  auto stats_of = [&](const _Float16* y, int C, long inner) {
    chan_stats<<<C, 256, 0, stream>>>(y, stats, C, inner);
  };
  auto bn = [&](_Float16* y, const float* g, const float* bt, int C, long inner, int act) {
    long total = (long)BATCH * C * inner;
    bn_act<<<(unsigned)((total + 255) / 256), 256, 0, stream>>>(
        y, stats, g, bt, C, inner, total, 1.f / (float)((long)BATCH * inner), act);
  };

  // ---- graph: KNN -> edge convs -> conv3 -----------------------------------
  knn_topk<<<(BATCH * NPTS) / 256, 256, 0, stream>>>(x, idx);
  edge_conv1<<<BATCH * NPTS, 64, 0, stream>>>(x, idx, c1_w, e1);
  stats_of(e1, 64, NKI);
  bn(e1, bn1_g, bn1_b, 64, NKI, 1);
  gemm(w_c2, e1, nullptr, e2, false, 64, 64, NKI);
  stats_of(e2, 64, NKI);
  bn_leaky_maxk<<<(BATCH * 64 * NPTS) / 256, 256, 0, stream>>>(e2, stats, bn2_g, bn2_b, h64);
  gemm(w_c3, h64, nullptr, h128, false, 128, 64, NPTS);
  stats_of(h128, 128, NPTS);
  bn(h128, bn3_g, bn3_b, 128, NPTS, 1);

  // ---- 4 self-attention layers --------------------------------------------
  long nh = (long)BATCH * 128 * NPTS;
  for (int i = 0; i < 4; i++) {
    gemm(w_qk + i * 4096,  h128, nullptr,         qk16, false, 32,  128, NPTS);
    gemm(w_vw + i * 16384, h128, sa_vb + i * 128, v16b, false, 128, 128, NPTS);
    attn_rowstats<<<(BATCH * NPTS) / 8, 256, 0, stream>>>(qk16, rmax, rrcp);
    attn_xr<<<dim3(NPTS / 16, BATCH), 128, 0, stream>>>(qk16, v16b, rmax, rrcp, xr16);
    sub_k<<<(unsigned)((nh + 255) / 256), 256, 0, stream>>>(h128, xr16, tmp16, nh);
    gemm(w_tw + i * 16384, tmp16, sa_tb + i * 128, y16, false, 128, 128, NPTS);
    stats_of(y16, 128, NPTS);
    bn(y16, sa_g + i * 128, sa_b + i * 128, 128, NPTS, 2);
    sa_update<<<(unsigned)((nh + 255) / 256), 256, 0, stream>>>(h128, y16, cat, i);
  }

  // ---- tail: conv4 -> pool -> conv5/6/7 ------------------------------------
  gemm(w_c4, cat, nullptr, f4, false, 1024, 512, NPTS);
  stats_of(f4, 1024, NPTS);
  bn(f4, bn4_g, bn4_b, 1024, NPTS, 1);
  pool_maxmean<<<BATCH * 1024, 256, 0, stream>>>(f4, pooled);
  long nc2 = (long)BATCH * 3072 * NPTS;
  build_cat2<<<(unsigned)((nc2 + 255) / 256), 256, 0, stream>>>(f4, pooled, cat2);
  gemm(w_c5, cat2, c5_b, f5, false, 512, 3072, NPTS);
  stats_of(f5, 512, NPTS);
  bn(f5, bn5_g, bn5_b, 512, NPTS, 2);
  gemm(w_c6, f5, c6_b, f6, false, 256, 512, NPTS);
  stats_of(f6, 256, NPTS);
  bn(f6, bn6_g, bn6_b, 256, NPTS, 2);
  gemm(w_c7, f6, c7_b, d_out, true, 6, 256, NPTS);
}
